// Message_76845554860742
// MI455X (gfx1250) — matrix operand order repaired and checked by reading, not compiled, not run on hardware
//
#include <hip/hip_runtime.h>
#include <hip/hip_bf16.h>

typedef __attribute__((ext_vector_type(16))) _Float16 v16h;
typedef __attribute__((ext_vector_type(8)))  _Float16 v8h;
typedef __attribute__((ext_vector_type(4)))  _Float16 v4h;
typedef __attribute__((ext_vector_type(8)))  float    v8f;

#define N_FEAT    64
#define N_HIDDEN  128
#define N_EFEAT   16
#define KP        160   // 144 padded to multiple of 32
#define K2        128

// ---- WMMA fragment loaders (LDS, f16) -------------------------------------
// A 16x32 (MxK), 16-bit: lanes 0-15 hold M=lane with K k0+[0..7] (v0-3) and
// k0+[16..23] (v4-7); lanes 16-31 hold the same rows with K k0+[8..15] and
// k0+[24..31].
__device__ __forceinline__ v16h load_a_frag(const _Float16* rowp, int k0, int half) {
    const _Float16* p = rowp + k0 + half * 8;
    v8h lo = *(const v8h*)(p);
    v8h hi = *(const v8h*)(p + 16);
    v16h a;
#pragma unroll
    for (int i = 0; i < 8; ++i) { a[i] = lo[i]; a[i + 8] = hi[i]; }
    return a;
}

// B 32x16 (KxN), 16-bit: lane n (0-15) holds column N=n with K k0+[0..7] and
// k0+[16..23]; lanes 16-31 hold K k0+[8..15] and k0+[24..31]. Weights are stored transposed
// (N-major, K contiguous) in LDS so each lane reads contiguous halves.
__device__ __forceinline__ v16h load_b_frag(const _Float16* colp, int k0, int half) {
    const _Float16* p = colp + k0 + half * 8;
    v8h lo = *(const v8h*)(p);
    v8h hi = *(const v8h*)(p + 16);
    v16h b;
#pragma unroll
    for (int i = 0; i < 8; ++i) { b[i] = lo[i]; b[i + 8] = hi[i]; }
    return b;
}

__device__ __forceinline__ void lds_fence() {
    asm volatile("s_wait_dscnt 0" ::: "memory");
}

// ---- zero workspace --------------------------------------------------------
__global__ void zero_kernel(float* __restrict__ p, long long n) {
    long long i = (long long)blockIdx.x * blockDim.x + threadIdx.x;
    if (i < n) p[i] = 0.0f;
}

// ---- main edge-MLP + scatter kernel ---------------------------------------
__global__ void __launch_bounds__(256)
msg_kernel(const float* __restrict__ x,
           const long long* __restrict__ ei,   // [2, E]
           const float* __restrict__ ef,       // [E, 16]
           const float* __restrict__ W1, const float* __restrict__ b1,
           const float* __restrict__ W2, const float* __restrict__ b2,
           const float* __restrict__ W3, const float* __restrict__ b3,
           float* __restrict__ sums,           // [nodes, 64]
           float* __restrict__ counts,         // [nodes]
           int E) {
    extern __shared__ char smem[];
    _Float16* w1t  = (_Float16*)smem;            // [128][160] W1^T, K padded
    _Float16* w2t  = w1t + N_HIDDEN * KP;        // [128][128] W2^T
    _Float16* w3t  = w2t + N_HIDDEN * K2;        // [ 64][128] W3^T
    float*    b1f  = (float*)(w3t + N_FEAT * K2);
    float*    b2f  = b1f + N_HIDDEN;
    float*    b3f  = b2f + N_HIDDEN;
    int*      etgt = (int*)(b3f + N_FEAT);       // [8 waves][16] target ids
    _Float16* abuf = (_Float16*)(etgt + 128);    // [8*16][160] A tiles / h2
    _Float16* hbuf = abuf + 128 * KP;            // [8*16][128] h1 tiles

    const int tid  = threadIdx.x;
    const int wave = tid >> 5;
    const int lane = tid & 31;
    const int half = lane >> 4;
    const int lr   = lane & 15;

    // ---- one-time cooperative weight load (transposed, f16) ----
    for (int i = tid; i < N_HIDDEN * KP; i += 256) {
        int n = i / KP, k = i % KP;
        w1t[i] = (k < 144) ? (_Float16)W1[k * N_HIDDEN + n] : (_Float16)0.0f;
    }
    for (int i = tid; i < N_HIDDEN * K2; i += 256) {
        int n = i >> 7, k = i & 127;
        w2t[i] = (_Float16)W2[k * N_HIDDEN + n];
    }
    for (int i = tid; i < N_FEAT * K2; i += 256) {
        int n = i >> 7, k = i & 127;
        w3t[i] = (_Float16)W3[k * N_FEAT + n];
    }
    for (int i = tid; i < N_HIDDEN; i += 256) { b1f[i] = b1[i]; b2f[i] = b2[i]; }
    for (int i = tid; i < N_FEAT; i += 256) b3f[i] = b3[i];
    __syncthreads();

    _Float16* aw = abuf + wave * 16 * KP;   // this wave's A tile (later h2)
    _Float16* hw = hbuf + wave * 16 * K2;   // this wave's h1 tile
    int*      et = etgt + wave * 16;

    const int nTiles = (E + 15) >> 4;
    const int gw  = blockIdx.x * 8 + wave;
    const int gws = gridDim.x * 8;

    for (int t = gw; t < nTiles; t += gws) {
        const int e0 = t << 4;

        // ---- build A tile: [x[src] | x[tgt] | ef | 0-pad] as f16 ----
        {
            int e  = e0 + lr;
            int ec = (e < E) ? e : (E - 1);
            long long s = ei[ec];
            long long g = ei[E + ec];
            long long r = (half == 0) ? s : g;
            const float4* sp = (const float4*)(x + r * N_FEAT);
            _Float16* dst = aw + lr * KP + half * N_FEAT;
#pragma unroll
            for (int i = 0; i < 16; ++i) {
                float4 v = sp[i];
                v4h h4 = {(_Float16)v.x, (_Float16)v.y, (_Float16)v.z, (_Float16)v.w};
                *(v4h*)(dst + i * 4) = h4;
            }
            if (half == 0) {
                et[lr] = (int)g;
                const float4* ep = (const float4*)(ef + (long long)ec * N_EFEAT);
                _Float16* d2 = aw + lr * KP + 2 * N_FEAT;
#pragma unroll
                for (int i = 0; i < 4; ++i) {
                    float4 v = ep[i];
                    v4h h4 = {(_Float16)v.x, (_Float16)v.y, (_Float16)v.z, (_Float16)v.w};
                    *(v4h*)(d2 + i * 4) = h4;
                }
                if (e < E) atomicAdd(counts + (int)g, 1.0f);
            } else {
                _Float16* d2 = aw + lr * KP + 144;
                v4h z = {};
#pragma unroll
                for (int i = 0; i < 4; ++i) *(v4h*)(d2 + i * 4) = z;
            }
        }
        lds_fence();

        // ---- layer 1: [16x160] @ W1t -> ReLU -> h1 [16x128] ----
        {
            // Hoist A fragments: loaded once, reused across all 8 N-tiles.
            v16h a1[5];
            const _Float16* rowp = aw + lr * KP;
#pragma unroll
            for (int kc = 0; kc < 5; ++kc) a1[kc] = load_a_frag(rowp, kc * 32, half);

            for (int nt = 0; nt < 8; ++nt) {
                const int col = nt * 16 + lr;
                const float bv = b1f[col];
                v8f c = {bv, bv, bv, bv, bv, bv, bv, bv};
                const _Float16* colp = w1t + col * KP;
#pragma unroll
                for (int kc = 0; kc < 5; ++kc) {
                    v16h b = load_b_frag(colp, kc * 32, half);
                    c = __builtin_amdgcn_wmma_f32_16x16x32_f16(
                            false, a1[kc], false, b, (short)0, c, false, false);
                }
#pragma unroll
                for (int r = 0; r < 8; ++r) {
                    float v = fmaxf(c[r], 0.0f);
                    hw[(r + half * 8) * K2 + col] = (_Float16)v;
                }
            }
        }
        lds_fence();

        // ---- layer 2: h1 [16x128] @ W2t -> ReLU -> h2 (reuse A buffer) ----
        _Float16* h2 = aw;   // 16*128 fits in the 16*160 A region
        {
            v16h a2[4];
            const _Float16* rowp = hw + lr * K2;
#pragma unroll
            for (int kc = 0; kc < 4; ++kc) a2[kc] = load_a_frag(rowp, kc * 32, half);

            for (int nt = 0; nt < 8; ++nt) {
                const int col = nt * 16 + lr;
                const float bv = b2f[col];
                v8f c = {bv, bv, bv, bv, bv, bv, bv, bv};
                const _Float16* colp = w2t + col * K2;
#pragma unroll
                for (int kc = 0; kc < 4; ++kc) {
                    v16h b = load_b_frag(colp, kc * 32, half);
                    c = __builtin_amdgcn_wmma_f32_16x16x32_f16(
                            false, a2[kc], false, b, (short)0, c, false, false);
                }
#pragma unroll
                for (int r = 0; r < 8; ++r) {
                    float v = fmaxf(c[r], 0.0f);
                    h2[(r + half * 8) * K2 + col] = (_Float16)v;
                }
            }
        }
        lds_fence();

        // ---- layer 3: h2 [16x128] @ W3t -> message [16x64], scatter ----
        {
            v16h a3[4];
            const _Float16* rowp = h2 + lr * K2;
#pragma unroll
            for (int kc = 0; kc < 4; ++kc) a3[kc] = load_a_frag(rowp, kc * 32, half);

            for (int nt = 0; nt < 4; ++nt) {
                const int col = nt * 16 + lr;
                const float bv = b3f[col];
                v8f c = {bv, bv, bv, bv, bv, bv, bv, bv};
                const _Float16* colp = w3t + col * K2;
#pragma unroll
                for (int kc = 0; kc < 4; ++kc) {
                    v16h b = load_b_frag(colp, kc * 32, half);
                    c = __builtin_amdgcn_wmma_f32_16x16x32_f16(
                            false, a3[kc], false, b, (short)0, c, false, false);
                }
#pragma unroll
                for (int r = 0; r < 8; ++r) {
                    const int m = r + half * 8;
                    if (e0 + m < E) {
                        atomicAdd(sums + (long long)et[m] * N_FEAT + col, c[r]);
                    }
                }
            }
        }
        lds_fence();
    }
}

// ---- finalize: out = x + sums / max(counts, 1) ----------------------------
__global__ void fin_kernel(const float* __restrict__ x,
                           const float* __restrict__ sums,
                           const float* __restrict__ counts,
                           float* __restrict__ out, int total) {
    int i = blockIdx.x * blockDim.x + threadIdx.x;
    if (i < total) {
        int node = i >> 6;
        float d = fmaxf(counts[node], 1.0f);
        out[i] = x[i] + sums[i] / d;
    }
}

extern "C" void kernel_launch(void* const* d_in, const int* in_sizes, int n_in,
                              void* d_out, int out_size, void* d_ws, size_t ws_size,
                              hipStream_t stream) {
    const float*     x  = (const float*)d_in[0];
    const long long* ei = (const long long*)d_in[1];
    const float*     ef = (const float*)d_in[2];
    const float*     W1 = (const float*)d_in[3];
    const float*     b1 = (const float*)d_in[4];
    const float*     W2 = (const float*)d_in[5];
    const float*     b2 = (const float*)d_in[6];
    const float*     W3 = (const float*)d_in[7];
    const float*     b3 = (const float*)d_in[8];

    const int nodes = in_sizes[0] / N_FEAT;       // 50000
    const int E     = in_sizes[1] / 2;            // 1250000

    float* sums   = (float*)d_ws;                 // [nodes*64]
    float* counts = sums + (size_t)nodes * N_FEAT;// [nodes]

    long long zn = (long long)nodes * (N_FEAT + 1);
    zero_kernel<<<(int)((zn + 255) / 256), 256, 0, stream>>>(sums, zn);

    const size_t SMEM =
        (N_HIDDEN * KP + N_HIDDEN * K2 + N_FEAT * K2) * sizeof(_Float16)  // weights
        + (2 * N_HIDDEN + N_FEAT) * sizeof(float)                         // biases
        + 128 * sizeof(int)                                               // targets
        + (128 * KP + 128 * K2) * sizeof(_Float16);                       // A + h1
    (void)hipFuncSetAttribute(reinterpret_cast<const void*>(&msg_kernel),
                              hipFuncAttributeMaxDynamicSharedMemorySize,
                              (int)SMEM);

    msg_kernel<<<1024, 256, SMEM, stream>>>(x, ei, ef, W1, b1, W2, b2, W3, b3,
                                            sums, counts, E);

    int total = nodes * N_FEAT;
    fin_kernel<<<(total + 255) / 256, 256, 0, stream>>>(x, sums, counts,
                                                        (float*)d_out, total);
}
